// TinyTransformerLM_35424890258127
// MI455X (gfx1250) — matrix operand order repaired and checked by reading, not compiled, not run on hardware
//
#include <hip/hip_runtime.h>

#define DEV_INLINE __device__ __forceinline__
#define AS1 __attribute__((address_space(1)))
#define AS3 __attribute__((address_space(3)))

// Probe results (round 2):
//   PRESENT: global_load_async_to_lds_b64/b128, global_store_async_from_lds_b128,
//            cluster_load_async_to_lds_b128, s_wait_asynccnt, s_wait_tensorcnt,
//            tensor_load_to_lds, ds_atomic_async_barrier_arrive_b64
//   MISSING: ds_load_tr16_b128, ds_load_tr8_b64, global_load_tr16_b128
// async_to_lds param 1 type: 'int __vector_size__(16) __device__ *'  (v4i AS1*)

#if __has_builtin(__builtin_amdgcn_global_load_async_to_lds_b128) && \
    __has_builtin(__builtin_amdgcn_s_wait_asynccnt)
#define HAVE_ASYNC_LDS 1
#else
#define HAVE_ASYNC_LDS 0
#warning "PROBE MISSING: async-to-LDS builtins unavailable, using sync fallback"
#endif

// Model dims (fixed by the reference)
constexpr int cV = 32000, cD = 512, cL = 4, cH = 8, cDFF = 2048, cS = 1024, cB = 4;
constexpr int cDK = 64;
constexpr int cM = cB * cS; // 4096 token rows

typedef __bf16 bf16;
typedef __attribute__((ext_vector_type(16))) __bf16 v16bf;
typedef __attribute__((ext_vector_type(8)))  float  v8f;
typedef __attribute__((ext_vector_type(4)))  unsigned int u32x4;
typedef int v4i __attribute__((vector_size(16)));

union BF16Frag { u32x4 u4[2]; v16bf v; };

DEV_INLINE v8f wmma_bf16(v16bf a, v16bf b, v8f c) {
  // D = A(16x32 bf16) x B(32x16 bf16) + C(16x16 f32)
  return __builtin_amdgcn_wmma_f32_16x16x32_bf16(false, a, false, b, (short)0, c,
                                                 false, false);
}

DEV_INLINE void async_b128(const bf16* gsrc, bf16* ldst) {
#if HAVE_ASYNC_LDS
  __builtin_amdgcn_global_load_async_to_lds_b128((AS1 v4i*)gsrc, (AS3 v4i*)ldst,
                                                 0, 0);
#else
  *(u32x4*)ldst = *(const u32x4*)gsrc;
#endif
}

DEV_INLINE float redmax16(float x) {
  x = fmaxf(x, __shfl_xor(x, 1, 16));
  x = fmaxf(x, __shfl_xor(x, 2, 16));
  x = fmaxf(x, __shfl_xor(x, 4, 16));
  x = fmaxf(x, __shfl_xor(x, 8, 16));
  return x;
}
DEV_INLINE float redsum16(float x) {
  x += __shfl_xor(x, 1, 16);
  x += __shfl_xor(x, 2, 16);
  x += __shfl_xor(x, 4, 16);
  x += __shfl_xor(x, 8, 16);
  return x;
}

// ---------------------------------------------------------------- utilities
__global__ void f32_to_bf16_kernel(const float* __restrict__ in,
                                   bf16* __restrict__ out, size_t n) {
  size_t i = (size_t)blockIdx.x * blockDim.x + threadIdx.x;
  size_t stride = (size_t)gridDim.x * blockDim.x;
  for (; i < n; i += stride) out[i] = (bf16)in[i];
}

__global__ void embed_kernel(const int* __restrict__ idx,
                             const float* __restrict__ tok,
                             const float* __restrict__ pos,
                             float* __restrict__ x) {
  int row = blockIdx.x;        // b*S + s
  int s = row & (cS - 1);
  int t = idx[row];
  const float* te = tok + (size_t)t * cD;
  const float* pe = pos + (size_t)s * cD;
  float* xr = x + (size_t)row * cD;
  for (int c = threadIdx.x; c < cD; c += blockDim.x) xr[c] = te[c] + pe[c];
}

// One wave per row: LayerNorm over D=512, emit bf16
__global__ __launch_bounds__(32)
void ln_bf16_kernel(const float* __restrict__ x, const float* __restrict__ w,
                    const float* __restrict__ b, bf16* __restrict__ out) {
  int row = blockIdx.x;
  int lane = threadIdx.x;
  const float* xr = x + (size_t)row * cD;
  float v[16];
  float s = 0.f;
#pragma unroll
  for (int j = 0; j < 16; j++) { v[j] = xr[lane + 32 * j]; s += v[j]; }
#pragma unroll
  for (int m = 16; m >= 1; m >>= 1) s += __shfl_xor(s, m, 32);
  float mu = s * (1.0f / cD);
  float var = 0.f;
#pragma unroll
  for (int j = 0; j < 16; j++) { float d = v[j] - mu; var += d * d; }
#pragma unroll
  for (int m = 16; m >= 1; m >>= 1) var += __shfl_xor(var, m, 32);
  var *= (1.0f / cD);
  float rstd = rsqrtf(var + 1e-5f);
  bf16* orow = out + (size_t)row * cD;
#pragma unroll
  for (int j = 0; j < 16; j++) {
    int c = lane + 32 * j;
    orow[c] = (bf16)((v[j] - mu) * rstd * w[c] + b[c]);
  }
}

// ---------------------------------------------------------------- GEMM
// C[M,N] = act(A[M,K] @ W[N,K]^T + bias [+ res]); A,W bf16; acc f32.
// Double-buffered LDS tiles filled with async global->LDS DMA (ASYNCcnt).
constexpr int BM = 128, BN = 128, BKk = 64;
constexpr int LDT = BKk + 8;  // +16B pad, keeps 16B alignment (144B rows)

template<bool OUT_BF16, bool RELU, bool RES, bool BIAS>
__global__ __launch_bounds__(256)
void gemm_kernel(const bf16* __restrict__ A, const bf16* __restrict__ W,
                 const float* __restrict__ bias, const float* __restrict__ res,
                 void* __restrict__ Cout, int M, int N, int Kd) {
  __shared__ bf16 As[2][BM][LDT];
  __shared__ bf16 Bs[2][BN][LDT];
  const int tid = threadIdx.x;
  const int lane = tid & 31, wave = tid >> 5;
  const int wm = wave & 3, wn = wave >> 2;   // 4 row-slabs of 32, 2 col-slabs of 64
  const int half = lane >> 4, ln = lane & 15;
  const int m0 = blockIdx.y * BM, n0 = blockIdx.x * BN;

  v8f zero = {};
  v8f acc[2][4];
#pragma unroll
  for (int i = 0; i < 2; i++)
#pragma unroll
    for (int j = 0; j < 4; j++) acc[i][j] = zero;

  const int lrow = tid >> 1;          // 128 rows, 2 threads per row
  const int lcol = (tid & 1) * 32;    // 32 bf16 (64B) per thread

  const bf16* gArow = A + (size_t)(m0 + lrow) * Kd + lcol;
  const bf16* gBrow = W + (size_t)(n0 + lrow) * Kd + lcol;

  auto stage = [&](int bufi, int k0) {
    const bf16* ga = gArow + k0;
    const bf16* gb = gBrow + k0;
    bf16* la = &As[bufi][lrow][lcol];
    bf16* lb = &Bs[bufi][lrow][lcol];
#pragma unroll
    for (int i = 0; i < 4; i++) {
      async_b128(ga + i * 8, la + i * 8);
      async_b128(gb + i * 8, lb + i * 8);
    }
#if !HAVE_ASYNC_LDS
    if (k0 + BKk < Kd) {             // prefetch next tile (global_prefetch_b8)
      __builtin_prefetch(ga + BKk, 0, 3);
      __builtin_prefetch(gb + BKk, 0, 3);
    }
#endif
  };

  const int nk = Kd / BKk;
  int cur = 0;
  stage(0, 0);
  for (int kt = 0; kt < nk; ++kt) {
    const bool hasNext = (kt + 1) < nk;
    if (hasNext) stage(cur ^ 1, (kt + 1) * BKk);
#if HAVE_ASYNC_LDS
    // async loads complete in order: <=8 outstanding means current tile landed
    if (hasNext) __builtin_amdgcn_s_wait_asynccnt(8);
    else         __builtin_amdgcn_s_wait_asynccnt(0);
#endif
    __syncthreads();
#pragma unroll
    for (int ks = 0; ks < 2; ks++) {
      BF16Frag af[2], bw[4];
#pragma unroll
      for (int ms = 0; ms < 2; ms++) {
        int m = wm * 32 + ms * 16 + ln;
        af[ms].u4[0] = *(const u32x4*)&As[cur][m][ks * 32 + half * 16];
        af[ms].u4[1] = *(const u32x4*)&As[cur][m][ks * 32 + half * 16 + 8];
      }
#pragma unroll
      for (int ns = 0; ns < 4; ns++) {
        int n = wn * 64 + ns * 16 + ln;
        bw[ns].u4[0] = *(const u32x4*)&Bs[cur][n][ks * 32 + half * 16];
        bw[ns].u4[1] = *(const u32x4*)&Bs[cur][n][ks * 32 + half * 16 + 8];
      }
#pragma unroll
      for (int ms = 0; ms < 2; ms++)
#pragma unroll
        for (int ns = 0; ns < 4; ns++)
          acc[ms][ns] = wmma_bf16(af[ms].v, bw[ns].v, acc[ms][ns]);
    }
    __syncthreads();
    cur ^= 1;
  }

#pragma unroll
  for (int ms = 0; ms < 2; ms++) {
#pragma unroll
    for (int ns = 0; ns < 4; ns++) {
      const int ng = n0 + wn * 64 + ns * 16 + ln;
      float bv = BIAS ? bias[ng] : 0.0f;
#pragma unroll
      for (int e = 0; e < 8; e++) {
        const int mg = m0 + wm * 32 + ms * 16 + e + 8 * half;
        float val = acc[ms][ns][e] + bv;
        if (RES) val += res[(size_t)mg * N + ng];
        if (RELU) val = fmaxf(val, 0.0f);
        if (OUT_BF16) ((bf16*)Cout)[(size_t)mg * N + ng] = (bf16)val;
        else          ((float*)Cout)[(size_t)mg * N + ng] = val;
      }
    }
  }
}

// ---------------------------------------------------------------- attention
// Flash attention: one wave per (b, h, 16-row q tile). kv chunks of 32.
__global__ __launch_bounds__(32)
void attn_kernel(const bf16* __restrict__ Q, const bf16* __restrict__ Kx,
                 const bf16* __restrict__ Vx, bf16* __restrict__ O) {
  __shared__ bf16 p_lds[16][40];  // 16x32 P tile, padded
  __shared__ bf16 v_lds[32][72];  // 32x64 V tile, padded

  const int id = blockIdx.x;
  const int qt = id & 63;          // S/16 tiles
  const int h  = (id >> 6) & 7;
  const int b  = id >> 9;
  const int q0 = qt * 16;
  const int lane = threadIdx.x;
  const int half = lane >> 4, ln = lane & 15;

  const size_t bs = (size_t)b * cS * cD;
  const bf16* qp = Q + bs + (size_t)h * cDK;
  const bf16* kp = Kx + bs + (size_t)h * cDK;
  const bf16* vp = Vx + bs + (size_t)h * cDK;

  // Q A-frags for the two K-steps over DK=64
  BF16Frag qa[2];
  {
    const bf16* qrow = qp + (size_t)(q0 + ln) * cD;
#pragma unroll
    for (int ks = 0; ks < 2; ks++) {
      qa[ks].u4[0] = *(const u32x4*)(qrow + ks * 32 + half * 16);
      qa[ks].u4[1] = *(const u32x4*)(qrow + ks * 32 + half * 16 + 8);
    }
  }

  v8f zero = {};
  v8f oacc[4];
#pragma unroll
  for (int i = 0; i < 4; i++) oacc[i] = zero;
  float run_max[8], run_sum[8];
#pragma unroll
  for (int e = 0; e < 8; e++) { run_max[e] = -3.0e38f; run_sum[e] = 0.f; }

  const int nchunks = (q0 + 15) / 32 + 1;  // causal: kv <= q0+15
  for (int c = 0; c < nchunks; ++c) {
    const int kv0 = c * 32;

    // stage V chunk [32 x 64] into LDS early (async DMA, one row per lane)
    {
      const bf16* vrow = vp + (size_t)(kv0 + lane) * cD;
      bf16* dst = &v_lds[lane][0];
#pragma unroll
      for (int i = 0; i < 8; i++) async_b128(vrow + i * 8, dst + i * 8);
    }

    // scores: S = Q @ K^T   (two 16-col kv sub-tiles)
    v8f s0 = zero, s1 = zero;
    {
      const bf16* krow = kp + (size_t)(kv0 + ln) * cD;
      BF16Frag kb;
#pragma unroll
      for (int ks = 0; ks < 2; ks++) {
        kb.u4[0] = *(const u32x4*)(krow + ks * 32 + half * 16);
        kb.u4[1] = *(const u32x4*)(krow + ks * 32 + half * 16 + 8);
        s0 = wmma_bf16(qa[ks].v, kb.v, s0);
      }
    }
    {
      const bf16* krow = kp + (size_t)(kv0 + 16 + ln) * cD;
      BF16Frag kb;
#pragma unroll
      for (int ks = 0; ks < 2; ks++) {
        kb.u4[0] = *(const u32x4*)(krow + ks * 32 + half * 16);
        kb.u4[1] = *(const u32x4*)(krow + ks * 32 + half * 16 + 8);
        s1 = wmma_bf16(qa[ks].v, kb.v, s1);
      }
    }

    // online softmax; C layout: element e -> row q0+e+8*half, col ln (+16)
#pragma unroll
    for (int e = 0; e < 8; e++) {
      const int mg = q0 + e + 8 * half;
      float a0 = s0[e] * 0.125f;  // 1/sqrt(64)
      float a1 = s1[e] * 0.125f;
      if (kv0 + ln > mg)      a0 = -1.0e30f;
      if (kv0 + 16 + ln > mg) a1 = -1.0e30f;
      float rm = redmax16(fmaxf(a0, a1));
      float nm = fmaxf(run_max[e], rm);
      float alpha = __expf(run_max[e] - nm);
      run_max[e] = nm;
      float p0 = __expf(a0 - nm);
      float p1 = __expf(a1 - nm);
      float rs = redsum16(p0 + p1);
      run_sum[e] = run_sum[e] * alpha + rs;
#pragma unroll
      for (int nn = 0; nn < 4; nn++) oacc[nn][e] *= alpha;
      p_lds[e + 8 * half][ln]      = (bf16)p0;
      p_lds[e + 8 * half][16 + ln] = (bf16)p1;
    }
#if HAVE_ASYNC_LDS
    __builtin_amdgcn_s_wait_asynccnt(0);
#endif
    __syncthreads();

    // O += P @ V
    BF16Frag pa;
    pa.u4[0] = *(const u32x4*)&p_lds[ln][half * 16];
    pa.u4[1] = *(const u32x4*)&p_lds[ln][half * 16 + 8];
#pragma unroll
    for (int nn = 0; nn < 4; nn++) {
      BF16Frag vb;
#pragma unroll
      for (int j = 0; j < 16; j++) vb.v[j] = v_lds[half * 16 + j][nn * 16 + ln];
      oacc[nn] = wmma_bf16(pa.v, vb.v, oacc[nn]);
    }
    __syncthreads();
  }

  bf16* op = O + bs + (size_t)h * cDK;
#pragma unroll
  for (int nn = 0; nn < 4; nn++)
#pragma unroll
    for (int e = 0; e < 8; e++) {
      const int mg = q0 + e + 8 * half;
      float val = oacc[nn][e] / run_sum[e];
      op[(size_t)mg * cD + nn * 16 + ln] = (bf16)val;
    }
}

// ---------------------------------------------------------------- host
template<bool OB, bool RELU, bool RES, bool BIAS>
static void launch_gemm(const bf16* A, const bf16* W, const float* bias,
                        const float* res, void* C, int M, int N, int Kd,
                        hipStream_t s) {
  dim3 g(N / BN, M / BM);
  gemm_kernel<OB, RELU, RES, BIAS><<<g, 256, 0, s>>>(A, W, bias, res, C, M, N, Kd);
}

extern "C" void kernel_launch(void* const* d_in, const int* in_sizes, int n_in,
                              void* d_out, int out_size, void* d_ws, size_t ws_size,
                              hipStream_t stream) {
  (void)in_sizes; (void)n_in; (void)out_size; (void)ws_size;
  const int*   idx  = (const int*)d_in[0];
  const float* tokw = (const float*)d_in[1];
  const float* posw = (const float*)d_in[2];
  const float* Wq   = (const float*)d_in[3];
  const float* bq   = (const float*)d_in[4];
  const float* Wk   = (const float*)d_in[5];
  const float* bk   = (const float*)d_in[6];
  const float* Wv   = (const float*)d_in[7];
  const float* bv   = (const float*)d_in[8];
  const float* Wo   = (const float*)d_in[9];
  const float* bo   = (const float*)d_in[10];
  const float* ln1w = (const float*)d_in[11];
  const float* ln1b = (const float*)d_in[12];
  const float* ln2w = (const float*)d_in[13];
  const float* ln2b = (const float*)d_in[14];
  const float* W1   = (const float*)d_in[15];
  const float* b1   = (const float*)d_in[16];
  const float* W2   = (const float*)d_in[17];
  const float* b2   = (const float*)d_in[18];
  const float* lnfw = (const float*)d_in[19];
  const float* lnfb = (const float*)d_in[20];

  // carve workspace
  char* p = (char*)d_ws;
  auto take = [&](size_t elems, size_t esz) -> void* {
    void* r = (void*)p;
    p += (elems * esz + 255) & ~(size_t)255;
    return r;
  };
  const size_t nDD  = (size_t)cL * cD * cD;
  const size_t nFD  = (size_t)cL * cDFF * cD;
  const size_t nEmb = (size_t)cV * cD;
  bf16* wq_b   = (bf16*)take(nDD, 2);
  bf16* wk_b   = (bf16*)take(nDD, 2);
  bf16* wv_b   = (bf16*)take(nDD, 2);
  bf16* wo_b   = (bf16*)take(nDD, 2);
  bf16* w1_b   = (bf16*)take(nFD, 2);
  bf16* w2_b   = (bf16*)take(nFD, 2);
  bf16* temb_b = (bf16*)take(nEmb, 2);
  float* xbuf  = (float*)take((size_t)cM * cD, 4);
  bf16*  hbuf  = (bf16*)take((size_t)cM * cD, 2);
  bf16*  qbuf  = (bf16*)take((size_t)cM * cD, 2);
  bf16*  kbuf  = (bf16*)take((size_t)cM * cD, 2);
  bf16*  vbuf  = (bf16*)take((size_t)cM * cD, 2);
  bf16*  obuf  = (bf16*)take((size_t)cM * cD, 2);
  bf16*  ffbuf = (bf16*)take((size_t)cM * cDFF, 2);

  auto cvt = [&](const float* src, bf16* dst, size_t n) {
    f32_to_bf16_kernel<<<2048, 256, 0, stream>>>(src, dst, n);
  };
  cvt(Wq, wq_b, nDD);
  cvt(Wk, wk_b, nDD);
  cvt(Wv, wv_b, nDD);
  cvt(Wo, wo_b, nDD);
  cvt(W1, w1_b, nFD);
  cvt(W2, w2_b, nFD);
  cvt(tokw, temb_b, nEmb);

  embed_kernel<<<cM, 128, 0, stream>>>(idx, tokw, posw, xbuf);

  for (int l = 0; l < cL; ++l) {
    const bf16* wq_l = wq_b + (size_t)l * cD * cD;
    const bf16* wk_l = wk_b + (size_t)l * cD * cD;
    const bf16* wv_l = wv_b + (size_t)l * cD * cD;
    const bf16* wo_l = wo_b + (size_t)l * cD * cD;
    const bf16* w1_l = w1_b + (size_t)l * cDFF * cD;
    const bf16* w2_l = w2_b + (size_t)l * cD * cDFF;

    // h = LN1(x)
    ln_bf16_kernel<<<cM, 32, 0, stream>>>(xbuf, ln1w + (size_t)l * cD,
                                          ln1b + (size_t)l * cD, hbuf);
    // q,k,v = h @ W^T + b
    launch_gemm<true, false, false, true>(hbuf, wq_l, bq + (size_t)l * cD, nullptr,
                                          qbuf, cM, cD, cD, stream);
    launch_gemm<true, false, false, true>(hbuf, wk_l, bk + (size_t)l * cD, nullptr,
                                          kbuf, cM, cD, cD, stream);
    launch_gemm<true, false, false, true>(hbuf, wv_l, bv + (size_t)l * cD, nullptr,
                                          vbuf, cM, cD, cD, stream);
    // o = softmax(q k^T / sqrt(dk)) v  (causal, flash)
    attn_kernel<<<cB * cH * (cS / 16), 32, 0, stream>>>(qbuf, kbuf, vbuf, obuf);
    // x = x + o @ Wo^T + bo
    launch_gemm<false, false, true, true>(obuf, wo_l, bo + (size_t)l * cD, xbuf,
                                          xbuf, cM, cD, cD, stream);
    // h = LN2(x); ff = relu(h @ W1^T + b1); x = x + ff @ W2^T + b2
    ln_bf16_kernel<<<cM, 32, 0, stream>>>(xbuf, ln2w + (size_t)l * cD,
                                          ln2b + (size_t)l * cD, hbuf);
    launch_gemm<true, true, false, true>(hbuf, w1_l, b1 + (size_t)l * cDFF, nullptr,
                                         ffbuf, cM, cDFF, cD, stream);
    launch_gemm<false, false, true, true>(ffbuf, w2_l, b2 + (size_t)l * cD, xbuf,
                                          xbuf, cM, cD, cDFF, stream);
  }

  // final LN + tied LM head: logits = LN_f(x) @ tok_emb^T
  ln_bf16_kernel<<<cM, 32, 0, stream>>>(xbuf, lnfw, lnfb, hbuf);
  launch_gemm<false, false, false, false>(hbuf, temb_b, nullptr, nullptr, d_out,
                                          cM, cV, cD, stream);
}
